// HaloNetEncoder_45595372814576
// MI455X (gfx1250) — matrix operand order repaired and checked by reading, not compiled, not run on hardware
//
#include <hip/hip_runtime.h>
#include <hip/hip_bf16.h>
#include <math.h>
#include <stdint.h>

typedef __bf16 bf16;
typedef __attribute__((ext_vector_type(16))) __bf16 v16bf;
typedef __attribute__((ext_vector_type(8)))  float  v8f;

#define NLAYER 2
#define HEADS  12
#define DMODEL 768
#define SEQ    2048
#define NBATCH 8
#define BLKS   128
#define HALOW  32
#define INNERD 3072
#define DHEAD  64
#define NBLK   16
#define WINW   192
#define SPAD   (SEQ + 2 * HALOW)      /* 2112 */
#define MROWS  (NBATCH * SEQ)         /* 16384 */

union FragAB { v16bf v; uint4 q[2]; };

__device__ __forceinline__ float gelu_exact(float x) {
    return 0.5f * x * (1.0f + erff(x * 0.70710678118654752f));
}

/* ---------------- elementwise helpers ---------------- */

__global__ void cast_f32_bf16_kernel(const float* __restrict__ in, bf16* __restrict__ out, size_t n4) {
    /* n4 = number of float4 groups */
    size_t i = (size_t)blockIdx.x * blockDim.x + threadIdx.x;
    size_t st = (size_t)gridDim.x * blockDim.x;
    const float4* in4 = (const float4*)in;
    for (; i < n4; i += st) {
        float4 v = in4[i];
        bf16* o = out + i * 4;
        o[0] = (bf16)v.x; o[1] = (bf16)v.y; o[2] = (bf16)v.z; o[3] = (bf16)v.w;
    }
}

__global__ void zero_b16_kernel(uint4* __restrict__ p, size_t n16) {
    /* n16 = number of 16-byte groups */
    size_t i = (size_t)blockIdx.x * blockDim.x + threadIdx.x;
    size_t st = (size_t)gridDim.x * blockDim.x;
    const uint4 z = {0u, 0u, 0u, 0u};
    for (; i < n16; i += st) p[i] = z;
}

/* W[K][N] f32 -> WT[N][K] bf16, 32x32 tiles */
__global__ void transpose_cast_kernel(const float* __restrict__ W, bf16* __restrict__ WT,
                                      int K, int N) {
    __shared__ float tile[32][33];
    const int k0 = blockIdx.x * 32, n0 = blockIdx.y * 32;
    const int tx = threadIdx.x, ty = threadIdx.y;   /* (32,8) */
#pragma unroll
    for (int i = 0; i < 32; i += 8)
        tile[ty + i][tx] = W[(size_t)(k0 + ty + i) * N + n0 + tx];
    __syncthreads();
#pragma unroll
    for (int i = 0; i < 32; i += 8)
        WT[(size_t)(n0 + ty + i) * K + k0 + tx] = (bf16)tile[tx][ty + i];
}

/* V [B][S][D] bf16 -> VT [B*D][SPAD] bf16 (interior only; halos pre-zeroed) */
__global__ void transpose_v_kernel(const bf16* __restrict__ v, bf16* __restrict__ vt) {
    __shared__ bf16 tile[32][33];
    const int s0 = blockIdx.x * 32, d0 = blockIdx.y * 32, b = blockIdx.z;
    const int tx = threadIdx.x, ty = threadIdx.y;   /* (32,8) */
#pragma unroll
    for (int i = 0; i < 32; i += 8)
        tile[ty + i][tx] = v[(size_t)(b * SEQ + s0 + ty + i) * DMODEL + d0 + tx];
    __syncthreads();
#pragma unroll
    for (int i = 0; i < 32; i += 8)
        vt[(size_t)(b * DMODEL + d0 + ty + i) * SPAD + HALOW + s0 + tx] = tile[tx][ty + i];
}

/* ---------------- WMMA GEMM: C = act(A @ BT^T + bias [+ res]) ----------------
   A  [M][K] bf16 row-major, BT [N][K] bf16 (pre-transposed weights).
   128x128 tile per block (256 threads = 8 waves, each wave 32x64).
   Tile staging uses gfx1250 GLOBAL_LOAD_ASYNC_TO_LDS_B128 (ASYNCcnt).        */
template<bool HAS_RES, bool HAS_OUTF, bool HAS_OUTB, int ACT, bool KVPAD, bool USE_ASYNC>
__global__ __launch_bounds__(256) void gemm_wmma_kernel(
    const bf16* __restrict__ A, const bf16* __restrict__ BT,
    const float* __restrict__ bias, const float* __restrict__ res,
    float* __restrict__ outF, bf16* __restrict__ outB,
    int Mdim, int Ndim, int Kdim)
{
    __shared__ bf16 As[128 * 64];
    __shared__ bf16 Bs[128 * 64];
    const int tid  = threadIdx.x;
    const int wave = tid >> 5, lane = tid & 31;
    const int ln = lane & 15, lg = lane >> 4;
    const int tile_m = blockIdx.y * 128;
    const int tile_n = blockIdx.x * 128;
    const int wm = (wave >> 1) * 32;
    const int wn = (wave & 1) * 64;

    v8f c[2][4];
#pragma unroll
    for (int im = 0; im < 2; ++im)
#pragma unroll
        for (int in = 0; in < 4; ++in)
#pragma unroll
            for (int e = 0; e < 8; ++e) c[im][in][e] = 0.0f;

    const int srow = tid >> 1;
    const int scol = (tid & 1) * 32;
    const unsigned ldsA = (unsigned)(uintptr_t)(As + srow * 64 + scol);
    const unsigned ldsB = (unsigned)(uintptr_t)(Bs + srow * 64 + scol);

    for (int k0 = 0; k0 < Kdim; k0 += 64) {
        __syncthreads();    /* previous iteration's fragment reads complete */
        if (USE_ASYNC) {
            unsigned long long ga = (unsigned long long)(A  + (size_t)(tile_m + srow) * Kdim + k0 + scol);
            unsigned long long gb = (unsigned long long)(BT + (size_t)(tile_n + srow) * Kdim + k0 + scol);
            asm volatile(
                "global_load_async_to_lds_b128 %0, %2, off\n\t"
                "global_load_async_to_lds_b128 %0, %2, off offset:16\n\t"
                "global_load_async_to_lds_b128 %0, %2, off offset:32\n\t"
                "global_load_async_to_lds_b128 %0, %2, off offset:48\n\t"
                "global_load_async_to_lds_b128 %1, %3, off\n\t"
                "global_load_async_to_lds_b128 %1, %3, off offset:16\n\t"
                "global_load_async_to_lds_b128 %1, %3, off offset:32\n\t"
                "global_load_async_to_lds_b128 %1, %3, off offset:48\n\t"
                "s_wait_asynccnt 0"
                :: "v"(ldsA), "v"(ldsB), "v"(ga), "v"(gb) : "memory");
        } else {
            const uint4* gA = (const uint4*)(A  + (size_t)(tile_m + srow) * Kdim + k0 + scol);
            const uint4* gB = (const uint4*)(BT + (size_t)(tile_n + srow) * Kdim + k0 + scol);
            uint4* sA = (uint4*)(As + srow * 64 + scol);
            uint4* sB = (uint4*)(Bs + srow * 64 + scol);
#pragma unroll
            for (int j = 0; j < 4; ++j) { sA[j] = gA[j]; sB[j] = gB[j]; }
        }
        __syncthreads();    /* staged tiles visible to all waves */
#pragma unroll
        for (int ks = 0; ks < 64; ks += 32) {
            FragAB a[2], bf[4];
#pragma unroll
            for (int im = 0; im < 2; ++im) {
                const bf16* base = As + (wm + 16 * im + ln) * 64 + ks;
                a[im].q[0] = *(const uint4*)(base + 8 * lg);
                a[im].q[1] = *(const uint4*)(base + 16 + 8 * lg);
            }
#pragma unroll
            for (int in = 0; in < 4; ++in) {
                const bf16* base = Bs + (wn + 16 * in + ln) * 64 + ks + 8 * lg;
                bf[in].q[0] = *(const uint4*)(base);
                bf[in].q[1] = *(const uint4*)(base + 16);
            }
#pragma unroll
            for (int im = 0; im < 2; ++im)
#pragma unroll
                for (int in = 0; in < 4; ++in)
                    c[im][in] = __builtin_amdgcn_wmma_f32_16x16x32_bf16(
                        false, a[im].v, false, bf[in].v, (short)0, c[im][in], false, false);
        }
    }

#pragma unroll
    for (int im = 0; im < 2; ++im) {
#pragma unroll
        for (int in = 0; in < 4; ++in) {
            const int col = tile_n + wn + 16 * in + ln;
            const float bv = bias[col];
#pragma unroll
            for (int e = 0; e < 8; ++e) {
                const int row = tile_m + wm + 16 * im + 8 * lg + e;
                float v = c[im][in][e] + bv;
                if (HAS_RES) v += res[(size_t)row * Ndim + col];
                if (ACT == 1) v = gelu_exact(v);
                if (HAS_OUTF) outF[(size_t)row * Ndim + col] = v;
                if (HAS_OUTB) {
                    int orow = row;
                    if (KVPAD) orow = (row >> 11) * SPAD + HALOW + (row & (SEQ - 1));
                    outB[(size_t)orow * Ndim + col] = (bf16)v;
                }
            }
        }
    }
}

/* ---------------- Halo attention: one (batch, block, head) per CTA ----------
   scores = Q(128x64) @ Kpad^T, masked softmax in registers, ctx = P @ V via VT.
   8 waves, each owns a 16-row query strip (24 + 24 WMMAs per wave).          */
__global__ __launch_bounds__(256) void halo_attention_kernel(
    const bf16* __restrict__ q,     /* [B,S,D]      */
    const bf16* __restrict__ kpad,  /* [B,SPAD,D]   */
    const bf16* __restrict__ vt,    /* [B*D, SPAD]  */
    bf16* __restrict__ ctx)         /* [B,S,D]      */
{
    __shared__ bf16 pstrip[8][16 * WINW];   /* 48 KB */
    const int tid  = threadIdx.x;
    const int wave = tid >> 5, lane = tid & 31;
    const int ln = lane & 15, lg = lane >> 4;
    const int nb = blockIdx.x, h = blockIdx.y, b = blockIdx.z;

    /* Q A-fragments: 16 rows x 64, two K-steps */
    FragAB qa[2];
    {
        const bf16* qb = q + (size_t)(b * SEQ + nb * BLKS + wave * 16 + ln) * DMODEL + h * DHEAD;
#pragma unroll
        for (int t = 0; t < 2; ++t) {
            qa[t].q[0] = *(const uint4*)(qb + 32 * t + 8 * lg);
            qa[t].q[1] = *(const uint4*)(qb + 32 * t + 16 + 8 * lg);
        }
    }

    v8f sc[12];
#pragma unroll
    for (int j = 0; j < 12; ++j)
#pragma unroll
        for (int e = 0; e < 8; ++e) sc[j][e] = 0.0f;

    /* scores: B-fragments are rows of padded K (contiguous in D) */
#pragma unroll
    for (int j = 0; j < 12; ++j) {
        const bf16* kb = kpad + (size_t)(b * SPAD + nb * BLKS + 16 * j + ln) * DMODEL + h * DHEAD;
#pragma unroll
        for (int t = 0; t < 2; ++t) {
            FragAB kf;
            kf.q[0] = *(const uint4*)(kb + 32 * t + 8 * lg);
            kf.q[1] = *(const uint4*)(kb + 32 * t + 16 + 8 * lg);
            sc[j] = __builtin_amdgcn_wmma_f32_16x16x32_bf16(
                false, qa[t].v, false, kf.v, (short)0, sc[j], false, false);
        }
    }

    /* scale + halo mask (per-lane column) */
#pragma unroll
    for (int j = 0; j < 12; ++j) {
        const int pos = nb * BLKS + 16 * j + ln - HALOW;
        const bool valid = (pos >= 0) && (pos < SEQ);
#pragma unroll
        for (int e = 0; e < 8; ++e)
            sc[j][e] = valid ? sc[j][e] * 0.125f : -1.0e30f;
    }

    /* softmax: each lane owns rows (8*lg + e), columns ln of each 16-tile;
       row reductions across the 16-lane half via shfl_xor               */
    float rmax[8], rinv[8];
#pragma unroll
    for (int e = 0; e < 8; ++e) {
        float m = -3.0e38f;
#pragma unroll
        for (int j = 0; j < 12; ++j) m = fmaxf(m, sc[j][e]);
#pragma unroll
        for (int s = 1; s < 16; s <<= 1) m = fmaxf(m, __shfl_xor(m, s, 32));
        rmax[e] = m;
    }
#pragma unroll
    for (int e = 0; e < 8; ++e) {
        float s = 0.0f;
#pragma unroll
        for (int j = 0; j < 12; ++j) {
            float ev = __expf(sc[j][e] - rmax[e]);
            sc[j][e] = ev;
            s += ev;
        }
#pragma unroll
        for (int t = 1; t < 16; t <<= 1) s += __shfl_xor(s, t, 32);
        rinv[e] = 1.0f / s;
    }

    /* spill P strip to LDS (re-shape C-layout -> A-layout) */
#pragma unroll
    for (int j = 0; j < 12; ++j)
#pragma unroll
        for (int e = 0; e < 8; ++e)
            pstrip[wave][(8 * lg + e) * WINW + 16 * j + ln] = (bf16)(sc[j][e] * rinv[e]);
    __syncthreads();

    /* ctx = P(16x192) @ Vwin(192x64); B-fragments are rows of VT (contiguous in S) */
    v8f o[4];
#pragma unroll
    for (int jn = 0; jn < 4; ++jn)
#pragma unroll
        for (int e = 0; e < 8; ++e) o[jn][e] = 0.0f;

#pragma unroll
    for (int kt = 0; kt < 6; ++kt) {
        FragAB pa;
        const bf16* pb = &pstrip[wave][ln * WINW + 32 * kt];
        pa.q[0] = *(const uint4*)(pb + 8 * lg);
        pa.q[1] = *(const uint4*)(pb + 16 + 8 * lg);
#pragma unroll
        for (int jn = 0; jn < 4; ++jn) {
            const bf16* vb = vt + (size_t)(b * DMODEL + h * DHEAD + 16 * jn + ln) * SPAD
                               + nb * BLKS + 32 * kt + 8 * lg;
            FragAB vf;
            vf.q[0] = *(const uint4*)(vb);
            vf.q[1] = *(const uint4*)(vb + 16);
            o[jn] = __builtin_amdgcn_wmma_f32_16x16x32_bf16(
                false, pa.v, false, vf.v, (short)0, o[jn], false, false);
        }
    }

#pragma unroll
    for (int jn = 0; jn < 4; ++jn)
#pragma unroll
        for (int e = 0; e < 8; ++e) {
            const int srow = nb * BLKS + wave * 16 + 8 * lg + e;
            ctx[(size_t)(b * SEQ + srow) * DMODEL + h * DHEAD + 16 * jn + ln] = (bf16)o[jn][e];
        }
}

/* ---------------- LayerNorm over D=768, one CTA per row ---------------- */
template<bool HAS_OUTB>
__global__ __launch_bounds__(256) void layernorm_kernel(
    const float* __restrict__ x, const float* __restrict__ gam, const float* __restrict__ bet,
    float* __restrict__ outF, bf16* __restrict__ outB)
{
    __shared__ float red[256];
    const int row = blockIdx.x;
    const float* xr = x + (size_t)row * DMODEL;
    float s = 0.0f;
    for (int i = threadIdx.x; i < DMODEL; i += 256) s += xr[i];
    red[threadIdx.x] = s; __syncthreads();
    for (int off = 128; off > 0; off >>= 1) {
        if (threadIdx.x < off) red[threadIdx.x] += red[threadIdx.x + off];
        __syncthreads();
    }
    const float mu = red[0] * (1.0f / DMODEL);
    __syncthreads();
    float v = 0.0f;
    for (int i = threadIdx.x; i < DMODEL; i += 256) { float d = xr[i] - mu; v += d * d; }
    red[threadIdx.x] = v; __syncthreads();
    for (int off = 128; off > 0; off >>= 1) {
        if (threadIdx.x < off) red[threadIdx.x] += red[threadIdx.x + off];
        __syncthreads();
    }
    const float inv = rsqrtf(red[0] * (1.0f / DMODEL) + 1e-12f);
    for (int i = threadIdx.x; i < DMODEL; i += 256) {
        float y = (xr[i] - mu) * inv * gam[i] + bet[i];
        outF[(size_t)row * DMODEL + i] = y;
        if (HAS_OUTB) outB[(size_t)row * DMODEL + i] = (bf16)y;
    }
}

/* ---------------- host launch ---------------- */

extern "C" void kernel_launch(void* const* d_in, const int* in_sizes, int n_in,
                              void* d_out, int out_size, void* d_ws, size_t ws_size,
                              hipStream_t stream)
{
    (void)in_sizes; (void)n_in; (void)out_size; (void)ws_size;
    const float* x    = (const float*)d_in[0];
    const float* Wq   = (const float*)d_in[1];
    const float* bq   = (const float*)d_in[2];
    const float* Wk   = (const float*)d_in[3];
    const float* bk   = (const float*)d_in[4];
    const float* Wv   = (const float*)d_in[5];
    const float* bv   = (const float*)d_in[6];
    const float* Wo   = (const float*)d_in[7];
    const float* bo   = (const float*)d_in[8];
    const float* ln1g = (const float*)d_in[9];
    const float* ln1b = (const float*)d_in[10];
    const float* W1   = (const float*)d_in[11];
    const float* b1   = (const float*)d_in[12];
    const float* W2   = (const float*)d_in[13];
    const float* b2   = (const float*)d_in[14];
    const float* ln2g = (const float*)d_in[15];
    const float* ln2b = (const float*)d_in[16];
    float* out = (float*)d_out;

    char* ws = (char*)d_ws;
    size_t off = 0;
    auto alloc = [&](size_t bytes) -> char* {
        char* p = ws + off;
        off += (bytes + 255) & ~(size_t)255;
        return p;
    };

    bf16 *wqT[NLAYER], *wkT[NLAYER], *wvT[NLAYER], *woT[NLAYER], *w1T[NLAYER], *w2T[NLAYER];
    for (int l = 0; l < NLAYER; ++l) {
        wqT[l] = (bf16*)alloc((size_t)DMODEL * DMODEL * 2);
        wkT[l] = (bf16*)alloc((size_t)DMODEL * DMODEL * 2);
        wvT[l] = (bf16*)alloc((size_t)DMODEL * DMODEL * 2);
        woT[l] = (bf16*)alloc((size_t)DMODEL * DMODEL * 2);
        w1T[l] = (bf16*)alloc((size_t)INNERD * DMODEL * 2);  /* [INNER][D]  */
        w2T[l] = (bf16*)alloc((size_t)DMODEL * INNERD * 2);  /* [D][INNER]  */
    }
    bf16* xb    = (bf16*)alloc((size_t)MROWS * DMODEL * 2);
    bf16* qb    = (bf16*)alloc((size_t)MROWS * DMODEL * 2);
    bf16* kpadb = (bf16*)alloc((size_t)NBATCH * SPAD * DMODEL * 2);
    bf16* vb    = (bf16*)alloc((size_t)MROWS * DMODEL * 2);
    bf16* vtb   = (bf16*)alloc((size_t)NBATCH * DMODEL * SPAD * 2);
    bf16* ctxb  = (bf16*)alloc((size_t)MROWS * DMODEL * 2);
    bf16* attnb = (bf16*)alloc((size_t)MROWS * DMODEL * 2);
    bf16* h1b   = (bf16*)alloc((size_t)MROWS * INNERD * 2);
    float* tmpf  = (float*)alloc((size_t)MROWS * DMODEL * 4);
    float* attnf = (float*)alloc((size_t)MROWS * DMODEL * 4);

    const dim3 tb32x8(32, 8);

    /* pre-transpose + cast all weights to bf16 [N][K] */
    for (int l = 0; l < NLAYER; ++l) {
        const size_t dd = (size_t)DMODEL * DMODEL, di = (size_t)DMODEL * INNERD;
        transpose_cast_kernel<<<dim3(DMODEL/32, DMODEL/32), tb32x8, 0, stream>>>(Wq + l*dd, wqT[l], DMODEL, DMODEL);
        transpose_cast_kernel<<<dim3(DMODEL/32, DMODEL/32), tb32x8, 0, stream>>>(Wk + l*dd, wkT[l], DMODEL, DMODEL);
        transpose_cast_kernel<<<dim3(DMODEL/32, DMODEL/32), tb32x8, 0, stream>>>(Wv + l*dd, wvT[l], DMODEL, DMODEL);
        transpose_cast_kernel<<<dim3(DMODEL/32, DMODEL/32), tb32x8, 0, stream>>>(Wo + l*dd, woT[l], DMODEL, DMODEL);
        transpose_cast_kernel<<<dim3(DMODEL/32, INNERD/32), tb32x8, 0, stream>>>(W1 + l*di, w1T[l], DMODEL, INNERD);
        transpose_cast_kernel<<<dim3(INNERD/32, DMODEL/32), tb32x8, 0, stream>>>(W2 + l*di, w2T[l], INNERD, DMODEL);
    }

    for (int l = 0; l < NLAYER; ++l) {
        const float* hin = (l == 0) ? x : (out + (size_t)(l - 1) * MROWS * DMODEL);

        cast_f32_bf16_kernel<<<2048, 256, 0, stream>>>(hin, xb, (size_t)MROWS * DMODEL / 4);
        zero_b16_kernel<<<2048, 256, 0, stream>>>((uint4*)kpadb, (size_t)NBATCH * SPAD * DMODEL / 8);
        zero_b16_kernel<<<2048, 256, 0, stream>>>((uint4*)vtb,   (size_t)NBATCH * DMODEL * SPAD / 8);

        const dim3 gD(DMODEL/128, MROWS/128);     /* (6,128)  */
        const dim3 gI(INNERD/128, MROWS/128);     /* (24,128) */

        /* Q, K (padded store), V projections */
        gemm_wmma_kernel<false,false,true,0,false,true><<<gD, 256, 0, stream>>>(
            xb, wqT[l], bq + l*DMODEL, nullptr, nullptr, qb, MROWS, DMODEL, DMODEL);
        gemm_wmma_kernel<false,false,true,0,true,true><<<gD, 256, 0, stream>>>(
            xb, wkT[l], bk + l*DMODEL, nullptr, nullptr, kpadb, MROWS, DMODEL, DMODEL);
        gemm_wmma_kernel<false,false,true,0,false,true><<<gD, 256, 0, stream>>>(
            xb, wvT[l], bv + l*DMODEL, nullptr, nullptr, vb, MROWS, DMODEL, DMODEL);

        transpose_v_kernel<<<dim3(SEQ/32, DMODEL/32, NBATCH), tb32x8, 0, stream>>>(vb, vtb);

        halo_attention_kernel<<<dim3(NBLK, HEADS, NBATCH), 256, 0, stream>>>(qb, kpadb, vtb, ctxb);

        /* output projection + residual -> LN1 */
        gemm_wmma_kernel<true,true,false,0,false,true><<<gD, 256, 0, stream>>>(
            ctxb, woT[l], bo + l*DMODEL, hin, tmpf, nullptr, MROWS, DMODEL, DMODEL);
        layernorm_kernel<true><<<MROWS, 256, 0, stream>>>(
            tmpf, ln1g + l*DMODEL, ln1b + l*DMODEL, attnf, attnb);

        /* MLP: gelu(attn @ W1 + b1) @ W2 + b2 + attn -> LN2 */
        gemm_wmma_kernel<false,false,true,1,false,true><<<gI, 256, 0, stream>>>(
            attnb, w1T[l], b1 + l*INNERD, nullptr, nullptr, h1b, MROWS, INNERD, DMODEL);
        gemm_wmma_kernel<true,true,false,0,false,true><<<gD, 256, 0, stream>>>(
            h1b, w2T[l], b2 + l*DMODEL, attnf, tmpf, nullptr, MROWS, DMODEL, INNERD);
        layernorm_kernel<false><<<MROWS, 256, 0, stream>>>(
            tmpf, ln2g + l*DMODEL, ln2b + l*DMODEL, out + (size_t)l * MROWS * DMODEL, nullptr);
    }
}
